// Attention_4879082848940
// MI455X (gfx1250) — matrix operand order repaired and checked by reading, not compiled, not run on hardware
//
#include <hip/hip_runtime.h>
#include <hip/hip_bf16.h>

typedef __attribute__((ext_vector_type(16))) _Float16 v16h;
typedef __attribute__((ext_vector_type(8)))  _Float16 v8h;
typedef __attribute__((ext_vector_type(4)))  _Float16 v4h;
typedef __attribute__((ext_vector_type(8)))  float    v8f;

#define B_   2
#define SEQ  2048
#define DIM  1024
#define H_   16
#define DH   64
#define NEG_INF (-3.0e38f)

union AFrag { v16h v; v8h h[2]; };

// ---------------------------------------------------------------------------
// fp32 -> f16 weight conversion (vectorized by 4), optional scale
// ---------------------------------------------------------------------------
__global__ __launch_bounds__(256) void cvt_f16_kernel(const float* __restrict__ src,
                                                      _Float16* __restrict__ dst,
                                                      int n4, float scale) {
    int i = blockIdx.x * 256 + threadIdx.x;
    if (i < n4) {
        float4 s = ((const float4*)src)[i];
        v4h o;
        o[0] = (_Float16)(s.x * scale);
        o[1] = (_Float16)(s.y * scale);
        o[2] = (_Float16)(s.z * scale);
        o[3] = (_Float16)(s.w * scale);
        ((v4h*)dst)[i] = o;
    }
}

// ---------------------------------------------------------------------------
// RMSNorm-as-written: xn = x / max(||x||2, 1e-12) * sqrt(dim) * gamma  (f16 out)
// one block (256 threads) per row of 1024
// ---------------------------------------------------------------------------
__global__ __launch_bounds__(256) void rmsnorm_f16_kernel(const float* __restrict__ x,
                                                          const float* __restrict__ gamma,
                                                          _Float16* __restrict__ xn) {
    __shared__ float red[8];
    int row = blockIdx.x;
    int tid = threadIdx.x;
    float4 xv = ((const float4*)(x + (size_t)row * DIM))[tid];
    float ss = xv.x*xv.x + xv.y*xv.y + xv.z*xv.z + xv.w*xv.w;
    #pragma unroll
    for (int o = 16; o > 0; o >>= 1) ss += __shfl_xor(ss, o, 32);
    if ((tid & 31) == 0) red[tid >> 5] = ss;
    __syncthreads();
    float tot = 0.f;
    #pragma unroll
    for (int i = 0; i < 8; ++i) tot += red[i];
    float scale = 32.0f / fmaxf(sqrtf(tot), 1e-12f);   // sqrt(1024) == 32
    float4 gv = ((const float4*)gamma)[tid];
    v4h o;
    o[0] = (_Float16)(xv.x * scale * gv.x);
    o[1] = (_Float16)(xv.y * scale * gv.y);
    o[2] = (_Float16)(xv.z * scale * gv.z);
    o[3] = (_Float16)(xv.w * scale * gv.w);
    *(v4h*)(xn + (size_t)row * DIM + tid * 4) = o;
}

// ---------------------------------------------------------------------------
// Y[M,N] = A[M,K] * W[N,K]^T (torch Linear), f16 in.
// Wave tile 32(M) x 64(N): 2 A-fragments x 4 B-fragments -> 8 WMMAs per
// k-step, B reused across both M-subtiles (1.5 b128 loads per WMMA).
// Block = 8 waves arranged 2(M) x 4(N) -> block tile 64 x 256.
// storeMode: 0 = f32 row-major, 1 = f16 row-major,
//            2 = f16 V-transposed: out[(b*16+h)*64+d][j] (j contiguous/lane)
// ---------------------------------------------------------------------------
__global__ __launch_bounds__(256) void gemm_nt_f16(const _Float16* __restrict__ A,
                                                   const _Float16* __restrict__ W,
                                                   _Float16* __restrict__ outH,
                                                   float* __restrict__ outF,
                                                   int M, int N, int K, int storeMode) {
    int lane = threadIdx.x & 31;
    int wid  = threadIdx.x >> 5;
    int col  = lane & 15;
    int hfl  = lane >> 4;
    int m0 = blockIdx.x * 64  + (wid >> 2) * 32;
    int n0 = blockIdx.y * 256 + (wid & 3) * 64;

    v8f acc[2][4] = {};
    const _Float16* arow0 = A + (size_t)(m0 + col) * K;        // rows m0..m0+15
    const _Float16* arow1 = A + (size_t)(m0 + 16 + col) * K;   // rows m0+16..m0+31
    for (int kk = 0; kk < K; kk += 32) {
        AFrag a0, a1;
        a0.h[0] = *(const v8h*)(arow0 + kk + hfl * 16);
        a0.h[1] = *(const v8h*)(arow0 + kk + 8 + hfl * 16);
        a1.h[0] = *(const v8h*)(arow1 + kk + hfl * 16);
        a1.h[1] = *(const v8h*)(arow1 + kk + 8 + hfl * 16);
        #pragma unroll
        for (int c = 0; c < 4; ++c) {
            const _Float16* brow = W + (size_t)(n0 + c * 16 + col) * K + kk + hfl * 16;
            v16h b = *(const v16h*)brow;
            acc[0][c] = __builtin_amdgcn_wmma_f32_16x16x32_f16(
                false, a0.v, false, b, (short)0, acc[0][c], false, false);
            acc[1][c] = __builtin_amdgcn_wmma_f32_16x16x32_f16(
                false, a1.v, false, b, (short)0, acc[1][c], false, false);
        }
    }
    #pragma unroll
    for (int ms = 0; ms < 2; ++ms)
        #pragma unroll
        for (int c = 0; c < 4; ++c)
            #pragma unroll
            for (int v = 0; v < 8; ++v) {
                int row = m0 + ms * 16 + hfl * 8 + v;
                int n   = n0 + c * 16 + col;
                float val = acc[ms][c][v];
                if (storeMode == 0) {
                    outF[(size_t)row * N + n] = val;
                } else if (storeMode == 1) {
                    outH[(size_t)row * N + n] = (_Float16)val;
                } else {
                    // V store: row = b*SEQ + j, n = h*64 + d  ->  Vt[b,h,d,j]
                    int b = row >> 11, j = row & (SEQ - 1);
                    int hh = n >> 6,  d = n & 63;
                    outH[((size_t)(b * H_ + hh) * DH + d) * SEQ + j] = (_Float16)val;
                }
            }
}

// ---------------------------------------------------------------------------
// Causal flash attention. One wave per (b, h, 16-query tile). 4 waves/block,
// each with a private LDS slice. Keys processed 32 at a time; early exit past
// the diagonal. QK^T: 2 WMMAs; PV: 4 WMMAs with 16x64 fp32 accumulator.
// Kb: [b*n, h*dh] f16 (K rows). Vt: [b,h,d,n] f16 (pre-transposed V).
// ---------------------------------------------------------------------------
__global__ __launch_bounds__(128) void attn_kernel(const _Float16* __restrict__ Q,
                                                   const _Float16* __restrict__ Kb,
                                                   const _Float16* __restrict__ Vt,
                                                   const float* __restrict__ bias,
                                                   const unsigned char* __restrict__ mask,
                                                   _Float16* __restrict__ AO) {
    __shared__ __align__(16) float    sim[4][16][33];
    __shared__ __align__(16) _Float16 pb[4][16][32];
    __shared__ __align__(16) float    rbuf[4][16];

    int lane = threadIdx.x & 31;
    int w    = threadIdx.x >> 5;
    int col  = lane & 15;
    int hfl  = lane >> 4;
    int wflat = blockIdx.x * 4 + w;
    int mt = wflat & 127;
    int h  = (wflat >> 7) & 15;
    int bb = wflat >> 11;
    int m0 = mt * 16;

    // Q A-fragments for the two 32-wide d-steps (dh = 64)
    const _Float16* qrow = Q + (size_t)(bb * SEQ + m0 + col) * DIM + h * DH;
    AFrag qa[2];
    #pragma unroll
    for (int s = 0; s < 2; ++s) {
        qa[s].h[0] = *(const v8h*)(qrow + s * 32 + hfl * 16);
        qa[s].h[1] = *(const v8h*)(qrow + s * 32 + 8 + hfl * 16);
    }

    v8f oacc[4] = {};
    float mrow = NEG_INF;
    float lrow = 0.f;
    const int jend = m0 + 16;   // causal: keys j < jend
    const float* biasrow = bias + ((size_t)h * SEQ + m0 + hfl * 8) * SEQ + col;

    for (int j0 = 0; j0 < jend; j0 += 32) {
        // prefetch the dominant HBM stream (bias) and K one block ahead
        if (j0 + 32 < jend) {
            __builtin_prefetch(biasrow + j0 + 32, 0, 0);
            __builtin_prefetch(Kb + (size_t)(bb * SEQ + j0 + 32 + col) * DIM + h * DH, 0, 0);
        }
        // ---- scores for two 16-key subtiles ----
        #pragma unroll
        for (int t = 0; t < 2; ++t) {
            int jb = j0 + t * 16;
            if (jb < jend) {
                v8f s = {};
                #pragma unroll
                for (int ds = 0; ds < 2; ++ds) {
                    const _Float16* krow = Kb + (size_t)(bb * SEQ + jb + col) * DIM
                                              + h * DH + ds * 32 + hfl * 16;
                    v16h kf = *(const v16h*)krow;
                    s = __builtin_amdgcn_wmma_f32_16x16x32_f16(
                        false, qa[ds].v, false, kf, (short)0, s, false, false);
                }
                int j = jb + col;
                bool mok = mask[bb * SEQ + j] != 0;
                #pragma unroll
                for (int v = 0; v < 8; ++v) {
                    int i = m0 + hfl * 8 + v;
                    float sv = s[v] + bias[((size_t)h * SEQ + i) * SEQ + j];
                    if (j > i || !mok) sv = NEG_INF;
                    sim[w][hfl * 8 + v][t * 16 + col] = sv;
                }
            } else {
                #pragma unroll
                for (int v = 0; v < 8; ++v)
                    sim[w][hfl * 8 + v][t * 16 + col] = NEG_INF;
            }
        }
        __builtin_amdgcn_wave_barrier();

        // ---- online softmax: lanes L and L+16 jointly own row L ----
        float vals[16], pm = NEG_INF;
        int c0 = hfl * 16;
        #pragma unroll
        for (int k = 0; k < 16; ++k) {
            vals[k] = sim[w][col][c0 + k];
            pm = fmaxf(pm, vals[k]);
        }
        pm = fmaxf(pm, __shfl_xor(pm, 16, 32));
        float mnew  = fmaxf(mrow, pm);
        float alpha = __expf(mrow - mnew);
        float psum = 0.f;
        #pragma unroll
        for (int k = 0; k < 16; ++k) {
            float e = __expf(vals[k] - mnew);
            psum += e;
            pb[w][col][c0 + k] = (_Float16)e;
        }
        psum += __shfl_xor(psum, 16, 32);
        lrow = lrow * alpha + psum;
        mrow = mnew;
        if (lane < 16) rbuf[w][col] = alpha;
        __builtin_amdgcn_wave_barrier();

        float al[8];
        #pragma unroll
        for (int v = 0; v < 8; ++v) al[v] = rbuf[w][hfl * 8 + v];
        #pragma unroll
        for (int c = 0; c < 4; ++c)
            #pragma unroll
            for (int v = 0; v < 8; ++v) oacc[c][v] *= al[v];

        // ---- O += P(16x32) @ V(32x64), P re-read from LDS in A-frag layout ----
        AFrag pf;
        pf.h[0] = *(const v8h*)&pb[w][col][hfl * 16];
        pf.h[1] = *(const v8h*)&pb[w][col][8 + hfl * 16];
        #pragma unroll
        for (int c = 0; c < 4; ++c) {
            int d = c * 16 + col;
            const _Float16* vrow = Vt + ((size_t)(bb * H_ + h) * DH + d) * SEQ + j0 + hfl * 16;
            v16h vf = *(const v16h*)vrow;
            oacc[c] = __builtin_amdgcn_wmma_f32_16x16x32_f16(
                false, pf.v, false, vf, (short)0, oacc[c], false, false);
        }
        __builtin_amdgcn_wave_barrier();
    }

    if (lane < 16) rbuf[w][col] = lrow;
    __builtin_amdgcn_wave_barrier();
    float li[8];
    #pragma unroll
    for (int v = 0; v < 8; ++v) li[v] = 1.0f / rbuf[w][hfl * 8 + v];
    #pragma unroll
    for (int c = 0; c < 4; ++c)
        #pragma unroll
        for (int v = 0; v < 8; ++v) {
            int i = m0 + hfl * 8 + v;
            AO[(size_t)(bb * SEQ + i) * DIM + h * DH + c * 16 + col] =
                (_Float16)(oacc[c][v] * li[v]);
        }
}

// ---------------------------------------------------------------------------
extern "C" void kernel_launch(void* const* d_in, const int* in_sizes, int n_in,
                              void* d_out, int out_size, void* d_ws, size_t ws_size,
                              hipStream_t stream) {
    const float* x    = (const float*)d_in[0];
    const float* bias = (const float*)d_in[1];
    const unsigned char* mask = (const unsigned char*)d_in[2];
    const float* gamma = (const float*)d_in[3];
    const float* Wq   = (const float*)d_in[4];
    const float* Wkv  = (const float*)d_in[5];
    const float* Wo   = (const float*)d_in[6];
    float* out = (float*)d_out;

    _Float16* ws   = (_Float16*)d_ws;                    // halves
    _Float16* xn   = ws;                                 // 4M  [b*n, dim]
    _Float16* wq   = xn   + (size_t)4 * 1024 * 1024;     // 1M
    _Float16* wkv  = wq   + (size_t)1 * 1024 * 1024;     // 2M
    _Float16* wo   = wkv  + (size_t)2 * 1024 * 1024;     // 1M
    _Float16* q    = wo   + (size_t)1 * 1024 * 1024;     // 4M  [b*n, h*dh]
    _Float16* kbuf = q    + (size_t)4 * 1024 * 1024;     // 4M  [b*n, h*dh]
    _Float16* vt   = kbuf + (size_t)4 * 1024 * 1024;     // 4M  [b*h, dh, n]
    _Float16* ao   = vt   + (size_t)4 * 1024 * 1024;     // 4M  [b*n, h*dh]

    // weights -> f16 (fold q scale dh^-0.5 = 0.125 into Wq)
    cvt_f16_kernel<<<1024, 256, 0, stream>>>(Wq,  wq,  262144, 0.125f);
    cvt_f16_kernel<<<2048, 256, 0, stream>>>(Wkv, wkv, 524288, 1.0f);
    cvt_f16_kernel<<<1024, 256, 0, stream>>>(Wo,  wo,  262144, 1.0f);

    // RMSNorm -> f16
    rmsnorm_f16_kernel<<<B_ * SEQ, 256, 0, stream>>>(x, gamma, xn);

    // projections: Q, K (row-major f16), V (stored directly transposed)
    gemm_nt_f16<<<dim3(64, 4), 256, 0, stream>>>(xn, wq,  q,    nullptr, 4096, 1024, 1024, 1);
    gemm_nt_f16<<<dim3(64, 4), 256, 0, stream>>>(xn, wkv, kbuf, nullptr, 4096, 1024, 1024, 1);
    gemm_nt_f16<<<dim3(64, 4), 256, 0, stream>>>(xn, wkv + (size_t)1024 * 1024,
                                                 vt, nullptr, 4096, 1024, 1024, 2);

    // causal flash attention: 4096 waves, 4 waves/block
    attn_kernel<<<1024, 128, 0, stream>>>(q, kbuf, vt, bias, mask, ao);

    // output projection, fp32 out
    gemm_nt_f16<<<dim3(64, 4), 256, 0, stream>>>(ao, wo, nullptr, out, 4096, 1024, 1024, 0);
}
